// TGAE_Encoder_3066606649575
// MI455X (gfx1250) — matrix operand order repaired and checked by reading, not compiled, not run on hardware
//
#include <hip/hip_runtime.h>
#include <hip/hip_bf16.h>

// ---------------------------------------------------------------------------
// GIN encoder for gfx1250 (MI455X). All GEMMs run on v_wmma_f32_16x16x32_bf16
// (bf16 inputs converted+packed on-the-fly into LDS tiles, f32 accumulation).
// Edge aggregation uses f32 global atomics (L2-resident working set).
// ---------------------------------------------------------------------------

#define NNODES 50000
#define NEDGES 1600000
#define D_IN   256
#define HID    512
#define D_CAT  768     // D_IN + HID
#define D_MID  1024    // 2*HID
#define D_SUM  1536    // 3*HID
#define D_OUT  256
#define LN_EPS 1e-5f

typedef __attribute__((ext_vector_type(16))) __bf16 v16bf;
typedef __attribute__((ext_vector_type(8)))  float  v8f;

__device__ __forceinline__ unsigned short f2bf(float f) {
    unsigned int u = __float_as_uint(f);
    u += 0x7fffu + ((u >> 16) & 1u);          // round-to-nearest-even
    return (unsigned short)(u >> 16);
}
__device__ __forceinline__ unsigned int pack2bf(float lo, float hi) {
    return (unsigned int)f2bf(lo) | ((unsigned int)f2bf(hi) << 16);
}

// ---------------------------------------------------------------------------
// Tiled WMMA GEMM:  C[M x N] = act( (A [+ A2]) @ B + bias ), optional copy C2.
// Block = 256 threads (8 waves). Block tile 128x64, K-step 32.
// Waves arranged 4(M) x 2(N); each wave computes a 32x32 C tile
// (2 A frags x 2 B frags -> 4 WMMAs per K-step).
// LDS rows padded to 40 halfwords (80 B) so every fragment group is a 16-byte
// aligned run -> ds_load_b128 / ds_store_b128.
// A, A2 row-major f32 (lda);  B row-major f32 [K x N] (ldb).
// act: 0 = none, 1 = leaky(0.1).
// ---------------------------------------------------------------------------
#define BM 128
#define BN 64
#define BK 32
#define LAS  40   // padded LDS row stride in halfwords
#define LASU 20   // same, in dwords

union FragU {
    v16bf v;
    unsigned int u[8];
    uint4 q[2];
};

__launch_bounds__(256)
__global__ void gemm_wmma_bf16(const float* __restrict__ A,
                               const float* __restrict__ A2, int lda,
                               const float* __restrict__ B, int ldb,
                               const float* __restrict__ bias,
                               float* __restrict__ C, int ldc,
                               float* __restrict__ C2, int ldc2,
                               int M, int K, int act)
{
    __shared__ __align__(16) unsigned short lA[BM * LAS]; // row-major [m][k]
    __shared__ __align__(16) unsigned short lB[BN * LAS]; // col-major [n][k]
    unsigned int* lAu = (unsigned int*)lA;
    unsigned int* lBu = (unsigned int*)lB;

    const int tid  = threadIdx.x;
    const int wid  = tid >> 5;
    const int lane = tid & 31;
    const int widM = wid & 3;   // 0..3 -> 32-row strip
    const int widN = wid >> 2;  // 0..1 -> 32-col strip
    const int row0 = blockIdx.x * BM;
    const int col0 = blockIdx.y * BN;

    v8f acc[2][2] = {};

    const int m  = lane & 15;
    const int hi = lane >> 4;
    // A frag i: row = widM*32 + i*16 + m ; dword groups at +hi*4 and +hi*4+8
    const int aIdx0 = (widM * 32 + m) * LASU + hi * 4;
    // B frag j: col = widN*32 + j*16 + m ; dword groups at +hi*4 and +hi*4+8
    const int bIdx0 = (widN * 32 + m) * LASU + hi * 4;

    // staging maps
    const int am = tid >> 1;            // A: row in tile (2 threads/row)
    const int ak = (tid & 1) * 16;      //    halfword offset (0 or 16)
    const int kp = tid & 15;            // B: k-pair index (k = 2*kp)
    const int ng = tid >> 4;            //    n-group (n0 = 4*ng)

    for (int kk = 0; kk < K; kk += BK) {
        __syncthreads();
        // ---- stage A tile (128x32) as packed bf16; fuse A2 add ----
        {
            uint4 p0, p1;
            const int gr = row0 + am;
            if (gr < M) {
                const float4* s = (const float4*)(A + (size_t)gr * lda + kk + ak);
                float4 f0 = s[0], f1 = s[1], f2 = s[2], f3 = s[3];
                if (A2) {
                    const float4* s2 = (const float4*)(A2 + (size_t)gr * lda + kk + ak);
                    float4 g0 = s2[0], g1 = s2[1], g2 = s2[2], g3 = s2[3];
                    f0.x += g0.x; f0.y += g0.y; f0.z += g0.z; f0.w += g0.w;
                    f1.x += g1.x; f1.y += g1.y; f1.z += g1.z; f1.w += g1.w;
                    f2.x += g2.x; f2.y += g2.y; f2.z += g2.z; f2.w += g2.w;
                    f3.x += g3.x; f3.y += g3.y; f3.z += g3.z; f3.w += g3.w;
                }
                p0 = make_uint4(pack2bf(f0.x, f0.y), pack2bf(f0.z, f0.w),
                                pack2bf(f1.x, f1.y), pack2bf(f1.z, f1.w));
                p1 = make_uint4(pack2bf(f2.x, f2.y), pack2bf(f2.z, f2.w),
                                pack2bf(f3.x, f3.y), pack2bf(f3.z, f3.w));
            } else {
                p0 = make_uint4(0, 0, 0, 0);
                p1 = make_uint4(0, 0, 0, 0);
            }
            uint4* dst = (uint4*)&lA[am * LAS + ak];   // 16-byte aligned
            dst[0] = p0;
            dst[1] = p1;
        }
        // ---- stage B tile (32x64) transposed into LDS col-major [n][k] ----
        {
            const float4 r0 = *(const float4*)(B + (size_t)(kk + 2 * kp) * ldb + col0 + 4 * ng);
            const float4 r1 = *(const float4*)(B + (size_t)(kk + 2 * kp + 1) * ldb + col0 + 4 * ng);
            lBu[(4 * ng + 0) * LASU + kp] = pack2bf(r0.x, r1.x);
            lBu[(4 * ng + 1) * LASU + kp] = pack2bf(r0.y, r1.y);
            lBu[(4 * ng + 2) * LASU + kp] = pack2bf(r0.z, r1.z);
            lBu[(4 * ng + 3) * LASU + kp] = pack2bf(r0.w, r1.w);
        }
        __syncthreads();

        FragU af[2], bq[2];
        #pragma unroll
        for (int i = 0; i < 2; ++i) {
            const int ai = aIdx0 + i * 16 * LASU;
            af[i].q[0] = *(const uint4*)&lAu[ai];        // K = hi*8 + 0..7
            af[i].q[1] = *(const uint4*)&lAu[ai + 8];    // K = hi*8 + 16..23
        }
        #pragma unroll
        for (int j = 0; j < 2; ++j) {
            const int bi = bIdx0 + j * 16 * LASU;
            bq[j].q[0] = *(const uint4*)&lBu[bi];        // K = hi*8 + 0..7
            bq[j].q[1] = *(const uint4*)&lBu[bi + 8];    // K = hi*8 + 16..23
        }
        #pragma unroll
        for (int i = 0; i < 2; ++i)
            #pragma unroll
            for (int j = 0; j < 2; ++j)
                acc[i][j] = __builtin_amdgcn_wmma_f32_16x16x32_bf16(
                    false, af[i].v, false, bq[j].v, (short)0, acc[i][j], false, false);
    }

    // ---- epilogue: bias, optional leaky, dual-destination store ----
    #pragma unroll
    for (int i = 0; i < 2; ++i) {
        #pragma unroll
        for (int j = 0; j < 2; ++j) {
            const int col = col0 + widN * 32 + j * 16 + (lane & 15);
            const float bv = bias ? bias[col] : 0.0f;
            #pragma unroll
            for (int v = 0; v < 8; ++v) {
                const int row = row0 + widM * 32 + i * 16 + (lane >> 4) * 8 + v;
                if (row < M) {
                    float val = acc[i][j][v] + bv;
                    if (act == 1) val = val > 0.0f ? val : 0.1f * val;
                    C[(size_t)row * ldc + col] = val;
                    if (C2) C2[(size_t)row * ldc2 + col] = val;
                }
            }
        }
    }
}

// ---------------------------------------------------------------------------
// Row-wise LayerNorm(1024) + leaky, in place. One block (256 thr) per row.
// ---------------------------------------------------------------------------
__launch_bounds__(256)
__global__ void ln_leaky_1024(float* __restrict__ X,
                              const float* __restrict__ g,
                              const float* __restrict__ b)
{
    const int row = blockIdx.x;
    float* x = X + (size_t)row * D_MID;
    const int tid = threadIdx.x;

    float vals[4];
    float s = 0.0f, sq = 0.0f;
    #pragma unroll
    for (int j = 0; j < 4; ++j) {
        float v = x[tid + j * 256];
        vals[j] = v; s += v; sq += v * v;
    }
    __shared__ float rs[256], rq[256];
    rs[tid] = s; rq[tid] = sq;
    __syncthreads();
    for (int off = 128; off > 0; off >>= 1) {
        if (tid < off) { rs[tid] += rs[tid + off]; rq[tid] += rq[tid + off]; }
        __syncthreads();
    }
    const float mean = rs[0] * (1.0f / D_MID);
    const float var  = rq[0] * (1.0f / D_MID) - mean * mean;
    const float rstd = rsqrtf(var + LN_EPS);
    #pragma unroll
    for (int j = 0; j < 4; ++j) {
        const int c = tid + j * 256;
        float y = (vals[j] - mean) * rstd * g[c] + b[c];
        x[c] = y > 0.0f ? y : 0.1f * y;
    }
}

// ---------------------------------------------------------------------------
// Edge scatter-add: agg[dst] += xcat[src] (768 f32 per edge).
// One wave per edge; float4 gathers; relaxed agent-scope f32 atomics (L2).
// ---------------------------------------------------------------------------
__launch_bounds__(256)
__global__ void edge_scatter(const int* __restrict__ ei,
                             const float* __restrict__ xcat,
                             float* __restrict__ agg)
{
    const int wid  = threadIdx.x >> 5;
    const int lane = threadIdx.x & 31;
    const long e = (long)blockIdx.x * 8 + wid;
    if (e >= NEDGES) return;
    const int s = ei[e];
    const int d = ei[NEDGES + e];
    const float4* srcp = (const float4*)(xcat + (size_t)s * D_CAT);
    float* dstp = agg + (size_t)d * D_CAT;
    #pragma unroll
    for (int c = lane; c < D_CAT / 4; c += 32) {
        const float4 v = srcp[c];
        const int b = c * 4;
        __hip_atomic_fetch_add(dstp + b + 0, v.x, __ATOMIC_RELAXED, __HIP_MEMORY_SCOPE_AGENT);
        __hip_atomic_fetch_add(dstp + b + 1, v.y, __ATOMIC_RELAXED, __HIP_MEMORY_SCOPE_AGENT);
        __hip_atomic_fetch_add(dstp + b + 2, v.z, __ATOMIC_RELAXED, __HIP_MEMORY_SCOPE_AGENT);
        __hip_atomic_fetch_add(dstp + b + 3, v.w, __ATOMIC_RELAXED, __HIP_MEMORY_SCOPE_AGENT);
    }
}

__launch_bounds__(256)
__global__ void zero_f4(float4* __restrict__ p, long n4)
{
    long i = (long)blockIdx.x * 256 + threadIdx.x;
    const long stride = (long)gridDim.x * 256;
    const float4 z = make_float4(0.f, 0.f, 0.f, 0.f);
    for (; i < n4; i += stride) p[i] = z;
}

// Copy x [N x 256] into xcat columns 0..255 (stride 768), float4-wide.
__launch_bounds__(256)
__global__ void copy_x_into_xcat(const float4* __restrict__ x4,
                                 float4* __restrict__ xcat4)
{
    const long i = (long)blockIdx.x * 256 + threadIdx.x;   // over N*64 float4s
    if (i < (long)NNODES * 64) {
        const long r = i >> 6;       // node
        const long c = i & 63;       // float4 col within 256
        xcat4[r * (D_CAT / 4) + c] = x4[i];
    }
}

// ---------------------------------------------------------------------------
extern "C" void kernel_launch(void* const* d_in, const int* in_sizes, int n_in,
                              void* d_out, int out_size, void* d_ws, size_t ws_size,
                              hipStream_t stream)
{
    const float* x     = (const float*)d_in[0];
    const int*   ei    = (const int*)  d_in[1];
    const float* in_w  = (const float*)d_in[2];
    const float* in_b  = (const float*)d_in[3];
    const float* w1    = (const float*)d_in[4];   // [2,768,1024]
    const float* b1    = (const float*)d_in[5];   // [2,1024]
    const float* ln_g  = (const float*)d_in[6];   // [2,1024]
    const float* ln_b  = (const float*)d_in[7];   // [2,1024]
    const float* w2    = (const float*)d_in[8];   // [2,1024,1024]
    const float* b2    = (const float*)d_in[9];   // [2,1024]
    const float* w3    = (const float*)d_in[10];  // [2,1024,512]
    const float* b3    = (const float*)d_in[11];  // [2,512]
    const float* out_w = (const float*)d_in[12];  // [1536,256]
    const float* out_b = (const float*)d_in[13];  // [256]
    float* out = (float*)d_out;

    // --- carve workspace (f32 activations) ---
    char* ws = (char*)d_ws;
    float* xcat = (float*)ws;                    ws += (size_t)NNODES * D_CAT * 4;
    float* agg  = (float*)ws;                    ws += (size_t)NNODES * D_CAT * 4;
    float* mid1 = (float*)ws;                    ws += (size_t)NNODES * D_MID * 4;
    float* mid2 = (float*)ws;                    ws += (size_t)NNODES * D_MID * 4;
    float* hcat = (float*)ws;                    ws += (size_t)NNODES * D_SUM * 4;

    const int gridM = (NNODES + BM - 1) / BM;    // 391
    const dim3 blk(256);

    // xcat[:, :256] = x
    copy_x_into_xcat<<<((long)NNODES * 64 + 255) / 256, blk, 0, stream>>>(
        (const float4*)x, (float4*)xcat);

    // h0 = x @ in_w + in_b  -> hcat[:, 0:512] and xcat[:, 256:768]
    gemm_wmma_bf16<<<dim3(gridM, HID / BN), blk, 0, stream>>>(
        x, nullptr, D_IN, in_w, HID, in_b,
        hcat, D_SUM, xcat + D_IN, D_CAT, NNODES, D_IN, /*act=*/0);

    for (int i = 0; i < 2; ++i) {
        // agg = 0
        zero_f4<<<4096, blk, 0, stream>>>((float4*)agg, (long)NNODES * D_CAT / 4);
        // agg[dst] += xcat[src]
        edge_scatter<<<NEDGES / 8, blk, 0, stream>>>(ei, xcat, agg);
        // mid1 = (xcat + agg) @ w1[i] + b1[i]      (h_pre fused into A-load)
        gemm_wmma_bf16<<<dim3(gridM, D_MID / BN), blk, 0, stream>>>(
            xcat, agg, D_CAT, w1 + (size_t)i * D_CAT * D_MID, D_MID,
            b1 + (size_t)i * D_MID, mid1, D_MID, nullptr, 0,
            NNODES, D_CAT, /*act=*/0);
        // mid1 = leaky(layernorm(mid1) * g + b)
        ln_leaky_1024<<<NNODES, blk, 0, stream>>>(
            mid1, ln_g + (size_t)i * D_MID, ln_b + (size_t)i * D_MID);
        // mid2 = leaky(mid1 @ w2[i] + b2[i])
        gemm_wmma_bf16<<<dim3(gridM, D_MID / BN), blk, 0, stream>>>(
            mid1, nullptr, D_MID, w2 + (size_t)i * D_MID * D_MID, D_MID,
            b2 + (size_t)i * D_MID, mid2, D_MID, nullptr, 0,
            NNODES, D_MID, /*act=*/1);
        // h = mid2 @ w3[i] + b3[i] -> hcat[:, 512*(i+1):512*(i+2)] and xcat[:, 256:]
        gemm_wmma_bf16<<<dim3(gridM, HID / BN), blk, 0, stream>>>(
            mid2, nullptr, D_MID, w3 + (size_t)i * D_MID * HID, HID,
            b3 + (size_t)i * HID, hcat + (size_t)HID * (i + 1), D_SUM,
            xcat + D_IN, D_CAT, NNODES, D_MID, /*act=*/0);
    }

    // out = hcat @ out_w + out_b
    gemm_wmma_bf16<<<dim3(gridM, D_OUT / BN), blk, 0, stream>>>(
        hcat, nullptr, D_SUM, out_w, D_OUT, out_b,
        out, D_OUT, nullptr, 0, NNODES, D_SUM, /*act=*/0);

    (void)in_sizes; (void)n_in; (void)out_size; (void)ws_size;
}
